// PS_APM_Seq_38302518346401
// MI455X (gfx1250) — matrix operand order repaired and checked by reading, not compiled, not run on hardware
//
#include <hip/hip_runtime.h>
#include <hip/hip_bf16.h>
#include <math.h>

// ---------------------------------------------------------------------------
// Types & helpers
// ---------------------------------------------------------------------------
typedef unsigned short u16;
typedef __attribute__((ext_vector_type(16))) __bf16          v16bf;
typedef __attribute__((ext_vector_type(8)))  float           v8f;
typedef __attribute__((ext_vector_type(16))) unsigned short  us16;
typedef __attribute__((ext_vector_type(8)))  unsigned short  us8;

__device__ __forceinline__ u16 f2bf(float f) {
    unsigned u = __builtin_bit_cast(unsigned, f);
    unsigned r = u + 0x7FFFu + ((u >> 16) & 1u);   // round-to-nearest-even
    return (u16)(r >> 16);
}
__device__ __forceinline__ float bf2f(u16 h) {
    unsigned u = ((unsigned)h) << 16;
    return __builtin_bit_cast(float, u);
}
__device__ __forceinline__ float sigmoidf_(float x) { return 1.f / (1.f + __expf(-x)); }

#define WMMA_BF16(a, b, c) \
    __builtin_amdgcn_wmma_f32_16x16x32_bf16(false, (a), false, (b), (short)0, (c), false, false)

// --- CDNA5 async global->LDS DMA (ASYNCcnt path) ---------------------------
// Each active lane copies 16 bytes: LDS[lds_off] = MEM[gptr].
__device__ __forceinline__ void async_copy16(unsigned lds_off, const void* gptr) {
    asm volatile("global_load_async_to_lds_b128 %0, %1, off"
                 :: "v"(lds_off), "v"(gptr)
                 : "memory");
}
__device__ __forceinline__ void wait_async0() {
    asm volatile("s_wait_asynccnt 0" ::: "memory");
}
// Generic LDS pointer -> wave-relative LDS byte offset (aperture low bits).
__device__ __forceinline__ unsigned lds_off_of(const void* p) {
    return (unsigned)(unsigned long long)p;
}

// A fragment: 16x32 bf16, rows M = lane&15, contiguous K halves (same order as B).
// lane<16: K in kb+0..15; lane>=16: K in kb+16..31.
__device__ __forceinline__ v16bf load_a_frag(const u16* base, int ld, int row0, int kb, int lane) {
    int lm = lane & 15, hi = (lane >> 4) & 1;
    const u16* p = base + (size_t)(row0 + lm) * ld + kb + 16 * hi;
    us8 lo = *(const us8*)p;
    us8 h8 = *(const us8*)(p + 8);
    us16 c;
#pragma unroll
    for (int i = 0; i < 8; ++i) { c[i] = lo[i]; c[i + 8] = h8[i]; }
    return __builtin_bit_cast(v16bf, c);
}
// B fragment from row-major weight W (Nout x K): B[k][n] = W[n][k].
// lanes 0-15 hold K=kb+0..15 for col n=lane; lanes 16-31 hold K=kb+16..31.
__device__ __forceinline__ v16bf load_b_frag(const u16* W, int ldw, int ncol0, int kb, int lane) {
    int lm = lane & 15, hi = (lane >> 4) & 1;
    us16 c = *(const us16*)(W + (size_t)(ncol0 + lm) * ldw + kb + 16 * hi);
    return __builtin_bit_cast(v16bf, c);
}

// ---------------------------------------------------------------------------
// Problem constants
// ---------------------------------------------------------------------------
#define BB 32
#define NN 64
#define CC 8
#define TIN 256
#define TOUT 64
#define HH 128
#define DTp 16
#define DPp 128
#define DIN 144

// ---------------------------------------------------------------------------
// Kernel 1: spectral norm, A matrices, positional table
// ---------------------------------------------------------------------------
__global__ __launch_bounds__(256) void prep_scalar_kernel(
    const float* __restrict__ Wneigh, const float* __restrict__ S,
    const float* __restrict__ G, const int* __restrict__ phases,
    const float* __restrict__ tpw, const float* __restrict__ tpb,
    float* __restrict__ Wn, float* __restrict__ Afull, float* __restrict__ ut)
{
    __shared__ float su[128], sv[144], swv[128];
    __shared__ float snorm;
    __shared__ float rs_[256], gval[256], gsum[4];
    const int tid = threadIdx.x;

    if (tid < 128) su[tid] = 0.08838834764831845f;   // 1/sqrt(128)
    __syncthreads();
    for (int it = 0; it < 30; ++it) {
        if (tid < 144) {
            float a = 0.f;
            for (int k = 0; k < 128; ++k) a += su[k] * Wneigh[k * 144 + tid];
            sv[tid] = a;
        }
        __syncthreads();
        if (tid == 0) {
            float nn = 0.f;
            for (int j = 0; j < 144; ++j) nn += sv[j] * sv[j];
            snorm = 1.f / (sqrtf(nn) + 1e-12f);
        }
        __syncthreads();
        if (tid < 144) sv[tid] *= snorm;
        __syncthreads();
        if (tid < 128) {
            float a = 0.f;
            for (int j = 0; j < 144; ++j) a += Wneigh[tid * 144 + j] * sv[j];
            swv[tid] = a;
        }
        __syncthreads();
        if (tid == 0) {
            float nn = 0.f;
            for (int k = 0; k < 128; ++k) nn += swv[k] * swv[k];
            snorm = 1.f / (sqrtf(nn) + 1e-12f);
        }
        __syncthreads();
        if (tid < 128) su[tid] = swv[tid] * snorm;
        __syncthreads();
    }
    if (tid == 0) {
        float sg = 0.f;
        for (int k = 0; k < 128; ++k) sg += su[k] * swv[k];
        snorm = 1.f / sg;
    }
    __syncthreads();
    for (int i = tid; i < 128 * 144; i += 256) Wn[i] = Wneigh[i] * snorm;

    // positional table u(t,d) = tanh(t/255 * w_d + b_d)
    for (int i = tid; i < 256 * 16; i += 256) {
        int t = i >> 4, d = i & 15;
        float tv = (float)t * (1.f / 255.f);
        ut[i] = tanhf(tv * tpw[d] + tpb[d]);
    }

    // per-phase |S| row sums and softplus(G) normalization
    {
        int p = tid >> 6, i = tid & 63;
        float a = 0.f;
        for (int j = 0; j < 64; ++j)
            if (j != i) a += fabsf(S[(p * 64 + i) * 64 + j]);
        rs_[tid] = fmaxf(a, 1e-6f);
        gval[tid] = log1pf(__expf(G[p * 64 + i])) + 1e-6f;
    }
    __syncthreads();
    if (tid < 4) {
        float s = 0.f;
        for (int j = 0; j < 64; ++j) s += gval[tid * 64 + j];
        gsum[tid] = fmaxf(s, 1e-6f);
    }
    __syncthreads();
    for (size_t idx = tid; idx < (size_t)BB * 64 * 64; idx += 256) {
        int j = (int)(idx & 63);
        int i = (int)((idx >> 6) & 63);
        int b = (int)(idx >> 12);
        int p = phases[b];
        float sz = (i == j) ? 0.f : S[(p * 64 + i) * 64 + j];
        float a = sz / rs_[p * 64 + i];
        a *= gval[p * 64 + i] * (64.f / gsum[p]);
        Afull[idx] = a;
    }
}

// ---------------------------------------------------------------------------
// Kernel 2: fp32 -> bf16 weight conversion with K padding
// ---------------------------------------------------------------------------
__global__ __launch_bounds__(256) void prep_weights_kernel(
    const float* __restrict__ gWih, const float* __restrict__ gWhh,
    const float* __restrict__ cWih, const float* __restrict__ cWhh,
    const float* __restrict__ Wk, const float* __restrict__ Wv,
    const float* __restrict__ Wself, const float* __restrict__ Wn,
    u16* __restrict__ whh, u16* __restrict__ wih, u16* __restrict__ cwih,
    u16* __restrict__ cwhh, u16* __restrict__ wkv, u16* __restrict__ wcat)
{
    size_t idx = (size_t)blockIdx.x * 256 + threadIdx.x;
    const size_t n_whh  = (size_t)64 * 384 * 128;
    const size_t n_wih  = (size_t)64 * 384 * 32;
    const size_t n_cwih = (size_t)384 * 160;
    const size_t n_cwhh = (size_t)384 * 128;
    const size_t n_wkv  = (size_t)256 * 128;
    const size_t n_wcat = (size_t)256 * 160;

    if (idx < n_whh) { whh[idx] = f2bf(gWhh[idx]); return; }
    idx -= n_whh;
    if (idx < n_wih) {
        size_t g = idx >> 5; int k = (int)(idx & 31);
        wih[(g << 5) + k] = (k < 8) ? f2bf(gWih[g * 8 + k]) : (u16)0;
        return;
    }
    idx -= n_wih;
    if (idx < n_cwih) {
        size_t g = idx / 160; int k = (int)(idx % 160);
        cwih[idx] = (k < 136) ? f2bf(cWih[g * 136 + k]) : (u16)0;
        return;
    }
    idx -= n_cwih;
    if (idx < n_cwhh) { cwhh[idx] = f2bf(cWhh[idx]); return; }
    idx -= n_cwhh;
    if (idx < n_wkv) {
        size_t r = idx >> 7; int k = (int)(idx & 127);
        wkv[idx] = f2bf(r < 128 ? Wk[r * 128 + k] : Wv[(r - 128) * 128 + k]);
        return;
    }
    idx -= n_wkv;
    if (idx < n_wcat) {
        size_t r = idx / 160; int k = (int)(idx % 160);
        float v = 0.f;
        if (k < 144) v = (r < 128) ? Wself[r * 144 + k] : Wn[(r - 128) * 144 + k];
        wcat[idx] = f2bf(v);
        return;
    }
}

// ---------------------------------------------------------------------------
// Kernel 3: GRU encoder. One block per node n, 8 waves, WMMA per step.
// Node weights are DMA'd into LDS ONCE via async global->LDS (ASYNCcnt),
// so the 256-step recurrence reads weights from LDS only.
// Dynamic LDS: whh 98304 | wih 24576 | hbf 8192 | hf 16384 | x 512  = 147,968B
// ---------------------------------------------------------------------------
__global__ __launch_bounds__(256) void gru_kernel(
    const float* __restrict__ X, const float* __restrict__ bih_g,
    const float* __restrict__ bhh_g, const u16* __restrict__ wih,
    const u16* __restrict__ whh, u16* __restrict__ Hc)
{
    const int n = blockIdx.x;
    const int tid = threadIdx.x;
    const int w = tid >> 5;
    const int lane = tid & 31;
    const int lm = lane & 15;
    const int hi = (lane >> 4) & 1;

    extern __shared__ __attribute__((aligned(32))) char dsm[];
    u16*   sh_whh = (u16*)dsm;                   // 384*128 bf16
    u16*   sh_wih = (u16*)(dsm + 98304);         // 384*32  bf16
    u16*   sh_hbf = (u16*)(dsm + 122880);        // 32*128  bf16
    float* sh_hf  = (float*)(dsm + 131072);      // 32*128  f32
    u16*   sh_x   = (u16*)(dsm + 147456);        // 32*8    bf16

    const u16* wihn = wih + (size_t)n * 384 * 32;
    const u16* whhn = whh + (size_t)n * 384 * 128;

    // --- async DMA of this node's weights into LDS (16B per lane per op) ---
    {
        const unsigned whh_off = lds_off_of(sh_whh);
        const unsigned wih_off = lds_off_of(sh_wih);
        for (int c = tid; c < 7680; c += 256) {      // 6144 whh + 1536 wih chunks
            if (c < 6144) async_copy16(whh_off + c * 16, whhn + c * 8);
            else          async_copy16(wih_off + (c - 6144) * 16, wihn + (c - 6144) * 8);
        }
        wait_async0();
    }
    for (int i = tid; i < 32 * 128; i += 256) { sh_hbf[i] = 0; sh_hf[i] = 0.f; }
    __syncthreads();

    const int col = 16 * w + lm;
    const float b_ir = bih_g[n * 384 + col],       b_hr = bhh_g[n * 384 + col];
    const float b_iz = bih_g[n * 384 + 128 + col], b_hz = bhh_g[n * 384 + 128 + col];
    const float b_in = bih_g[n * 384 + 256 + col], b_hn = bhh_g[n * 384 + 256 + col];

    for (int t = 0; t < TIN; ++t) {
        { // stage x_t into LDS as bf16
            int b = tid >> 3, c = tid & 7;
            sh_x[tid] = f2bf(X[(((size_t)b * NN + n) * CC + c) * TIN + t]);
        }
        __syncthreads();

        // Build padded-K A fragments for x (K=0..7 real, rest zero)
        v16bf ax[2];
#pragma unroll
        for (int m = 0; m < 2; ++m) {
            us16 c16;
#pragma unroll
            for (int i = 0; i < 16; ++i) c16[i] = 0;
            if (hi == 0) {
                int row = m * 16 + lm;
#pragma unroll
                for (int i = 0; i < 8; ++i) c16[i] = sh_x[row * 8 + i];
            }
            ax[m] = __builtin_bit_cast(v16bf, c16);
        }

        v8f ai[2][3], ah[2][3];
#pragma unroll
        for (int m = 0; m < 2; ++m)
#pragma unroll
            for (int g = 0; g < 3; ++g)
#pragma unroll
                for (int q = 0; q < 8; ++q) { ai[m][g][q] = 0.f; ah[m][g][q] = 0.f; }

        // gi = x @ Wih^T   (K = 32, one WMMA k-step, weights from LDS)
#pragma unroll
        for (int g = 0; g < 3; ++g) {
            v16bf bfr = load_b_frag(sh_wih, 32, g * 128 + 16 * w, 0, lane);
#pragma unroll
            for (int m = 0; m < 2; ++m)
                ai[m][g] = WMMA_BF16(ax[m], bfr, ai[m][g]);
        }
        // gh = h @ Whh^T   (K = 128, four WMMA k-steps, weights from LDS)
#pragma unroll
        for (int kb = 0; kb < 4; ++kb) {
            v16bf a0 = load_a_frag(sh_hbf, 128, 0, kb * 32, lane);
            v16bf a1 = load_a_frag(sh_hbf, 128, 16, kb * 32, lane);
#pragma unroll
            for (int g = 0; g < 3; ++g) {
                v16bf bfr = load_b_frag(sh_whh, 128, g * 128 + 16 * w, kb * 32, lane);
                ah[0][g] = WMMA_BF16(a0, bfr, ah[0][g]);
                ah[1][g] = WMMA_BF16(a1, bfr, ah[1][g]);
            }
        }
        __syncthreads();   // all reads of sh_hbf done before overwrite

        // GRU gates + state update + Hc store
#pragma unroll
        for (int m = 0; m < 2; ++m) {
#pragma unroll
            for (int v = 0; v < 8; ++v) {
                int row = m * 16 + v + 8 * hi;  // batch index
                float hold = sh_hf[row * 128 + col];
                float r  = sigmoidf_(ai[m][0][v] + b_ir + ah[m][0][v] + b_hr);
                float zg = sigmoidf_(ai[m][1][v] + b_iz + ah[m][1][v] + b_hz);
                float nn2 = tanhf(ai[m][2][v] + b_in + r * (ah[m][2][v] + b_hn));
                float hnew = (1.f - zg) * nn2 + zg * hold;
                sh_hf[row * 128 + col] = hnew;
                sh_hbf[row * 128 + col] = f2bf(hnew);
                Hc[(((size_t)row * NN + n) * TIN + t) * 160 + col] = f2bf(hnew);
            }
        }
        __syncthreads();
    }
}

// ---------------------------------------------------------------------------
// Kernel 4: LayerNorm over H in place + positional concat + K-padding
// ---------------------------------------------------------------------------
__global__ __launch_bounds__(256) void ln_kernel(
    u16* __restrict__ Hc, const float* __restrict__ ln_g,
    const float* __restrict__ ln_b, const float* __restrict__ ut)
{
    const int warp = threadIdx.x >> 5, lane = threadIdx.x & 31;
    const size_t row = (size_t)blockIdx.x * 8 + warp;
    u16* p = Hc + row * 160;
    float x[4];
    float s = 0.f, sq = 0.f;
#pragma unroll
    for (int q = 0; q < 4; ++q) {
        x[q] = bf2f(p[lane + 32 * q]);
        s += x[q]; sq += x[q] * x[q];
    }
    for (int m = 16; m >= 1; m >>= 1) {
        s  += __shfl_xor(s, m, 32);
        sq += __shfl_xor(sq, m, 32);
    }
    float mu = s * (1.f / 128.f);
    float var = sq * (1.f / 128.f) - mu * mu;
    float rs = rsqrtf(var + 1e-5f);
#pragma unroll
    for (int q = 0; q < 4; ++q) {
        int c = lane + 32 * q;
        p[c] = f2bf((x[q] - mu) * rs * ln_g[c] + ln_b[c]);
    }
    int t = (int)(row & 255);
    if (lane < 16) p[128 + lane] = f2bf(ut[t * 16 + lane]);
    else           p[144 + (lane - 16)] = 0;
}

// ---------------------------------------------------------------------------
// Kernel 5: generic tiled WMMA GEMM: out[M][N] = A[M][K] @ W[N][K]^T (bf16)
// block = 32 rows x 128 cols, 8 waves. The 32xK A tile is DMA'd into LDS
// once (async global->LDS) and shared by all 8 waves.
// Requires lda == K (true for both call sites), K <= 160, K % 32 == 0.
// ---------------------------------------------------------------------------
__global__ __launch_bounds__(256) void gemm_kernel(
    const u16* __restrict__ A, int lda, const u16* __restrict__ W, int ldw,
    u16* __restrict__ out, int ldo, int K)
{
    const int tid = threadIdx.x;
    const int w = tid >> 5, lane = tid & 31;
    const int lm = lane & 15, hi = (lane >> 4) & 1;
    const size_t rowbase = (size_t)blockIdx.x * 32;
    const int ncol0 = blockIdx.y * 128 + 16 * w;

    __shared__ __attribute__((aligned(32))) u16 sh_a[32 * 160];

    { // async DMA of the 32xK A tile into LDS (16B chunks)
        const unsigned abase = lds_off_of(sh_a);
        const int cpr = K >> 3;           // 16B chunks per row
        const int nch = 4 * K;            // total chunks (32 rows)
        for (int c = tid; c < nch; c += 256) {
            int row = c / cpr, off = c - row * cpr;
            async_copy16(abase + (unsigned)(row * K + off * 8) * 2,
                         A + (rowbase + row) * (size_t)lda + off * 8);
        }
        wait_async0();
    }
    __syncthreads();

    v8f acc0, acc1;
#pragma unroll
    for (int q = 0; q < 8; ++q) { acc0[q] = 0.f; acc1[q] = 0.f; }

    for (int kb = 0; kb < K; kb += 32) {
        v16bf a0 = load_a_frag(sh_a, K, 0, kb, lane);
        v16bf a1 = load_a_frag(sh_a, K, 16, kb, lane);
        v16bf bf = load_b_frag(W, ldw, ncol0, kb, lane);
        acc0 = WMMA_BF16(a0, bf, acc0);
        acc1 = WMMA_BF16(a1, bf, acc1);
    }
#pragma unroll
    for (int v = 0; v < 8; ++v) {
        size_t r0 = rowbase + v + 8 * hi;
        out[r0 * ldo + ncol0 + lm]        = f2bf(acc0[v]);
        out[(r0 + 16) * ldo + ncol0 + lm] = f2bf(acc1[v]);
    }
}

// ---------------------------------------------------------------------------
// Kernel 6: Z = leaky_relu(Hs + A @ Hn), extract h0 at t = T-1
// ---------------------------------------------------------------------------
__global__ __launch_bounds__(256) void mix_kernel(
    const u16* __restrict__ hshn, const float* __restrict__ Afull,
    u16* __restrict__ M, float* __restrict__ h0g)
{
    size_t idx = (size_t)blockIdx.x * 256 + threadIdx.x;
    int o = (int)(idx & 127);
    size_t rem = idx >> 7;
    int t = (int)(rem & 255);
    size_t rem2 = rem >> 8;
    int i = (int)(rem2 & 63);
    int b = (int)(rem2 >> 6);

    float hs = bf2f(hshn[((((size_t)b * NN + i) * TIN + t)) * 256 + o]);
    const float* Ar = Afull + ((size_t)b * NN + i) * 64;
    float acc = 0.f;
    for (int j = 0; j < 64; ++j) {
        size_t rj = ((size_t)b * NN + j) * TIN + t;
        acc += Ar[j] * bf2f(hshn[rj * 256 + 128 + o]);
    }
    float z = hs + acc;
    z = (z > 0.f) ? z : 0.1f * z;
    size_t mrow = ((size_t)b * NN + i) * TIN + t;
    M[mrow * 128 + o] = f2bf(z);
    if (t == TIN - 1) h0g[((size_t)b * NN + i) * 128 + o] = z;
}

// ---------------------------------------------------------------------------
// Kernel 7: decoder. One block per batch, 1024 threads (32 waves).
// Attention as 16-threads-per-row GEMV (with global_prefetch of next rows);
// GRU cell as WMMA (M=64, shared weights).
// ---------------------------------------------------------------------------
__global__ __launch_bounds__(1024) void decoder_kernel(
    const float* __restrict__ X, const float* __restrict__ Afull,
    const float* __restrict__ h0g, const u16* __restrict__ kv,
    const u16* __restrict__ cwih, const u16* __restrict__ cwhh,
    const float* __restrict__ cbih, const float* __restrict__ cbhh,
    const float* __restrict__ readW, const float* __restrict__ readb,
    const float* __restrict__ ltau, float* __restrict__ out)
{
    const int b = blockIdx.x;
    const int tid = threadIdx.x;
    const int wv = tid >> 5;
    const int lane = tid & 31;
    const int lm = lane & 15;
    const int hi = (lane >> 4) & 1;
    const int mt = wv >> 3;   // M-tile 0..3 (node rows)
    const int cg = wv & 7;    // col group 0..7
    const float scale = __expf(ltau[0]);

    extern __shared__ __attribute__((aligned(32))) char smem[];
    float* sh_hf    = (float*)smem;                       // 64*128 f32
    u16*   sh_hbf   = (u16*)(sh_hf + 64 * 128);           // 64*128 bf16
    u16*   sh_inp   = (u16*)(sh_hbf + 64 * 128);          // 64*160 bf16
    float* sh_alpha = (float*)(sh_inp + 64 * 160);        // 64*256 f32
    float* sh_ycur  = sh_alpha + 64 * 256;                // 64*8
    float* sh_yin   = sh_ycur + 512;                      // 64*8
    float* sh_xlast = sh_yin + 512;                       // 64*8

    for (int i = tid; i < 64 * 128; i += 1024) {
        float h = h0g[(size_t)b * 64 * 128 + i];
        sh_hf[i] = h; sh_hbf[i] = f2bf(h);
    }
    for (int i = tid; i < 64 * 160; i += 1024) sh_inp[i] = 0;
    if (tid < 512) {
        int i = tid >> 3, c = tid & 7;
        sh_ycur[tid] = 0.f;
        sh_xlast[tid] = X[(((size_t)b * NN + i) * CC + c) * TIN + (TIN - 1)];
    }
    __syncthreads();

    const int col = 16 * cg + lm;
    const float b_ir = cbih[col],       b_hr = cbhh[col];
    const float b_iz = cbih[128 + col], b_hz = cbhh[128 + col];
    const float b_in = cbih[256 + col], b_hn = cbhh[256 + col];
    const float* Ab = Afull + (size_t)b * 4096;

    for (int t = 0; t < TOUT; ++t) {
        // 1. graph-mixed feedback -> y_in (also into inp cols 128..135)
        if (tid < 512) {
            int i = tid >> 3, c = tid & 7;
            float acc = 0.f;
            for (int j = 0; j < 64; ++j) acc += Ab[i * 64 + j] * sh_ycur[j * 8 + c];
            float yin = sh_ycur[i * 8 + c] + 0.3f * acc;
            sh_yin[tid] = yin;
            sh_inp[i * 160 + 128 + c] = f2bf(yin);
        }
        __syncthreads();

        // 2+3. attention: 16 threads per row r
        {
            int r = tid >> 4;
            int l16 = tid & 15;
            const u16* kr = kv + ((size_t)(b * NN + r)) * TIN * 256;
            float sc[16];
            float mx = -1e30f;
#pragma unroll
            for (int ii = 0; ii < 16; ++ii) {
                int tq = l16 + 16 * ii;
                const u16* kp = kr + (size_t)tq * 256;
                if (ii < 15) __builtin_prefetch(kp + 16 * 256, 0, 0);  // next K row
                float s = 0.f;
                for (int kk = 0; kk < 128; ++kk) s += bf2f(kp[kk]) * sh_hf[r * 128 + kk];
                sc[ii] = s * scale;
                mx = fmaxf(mx, sc[ii]);
            }
            for (int m = 8; m >= 1; m >>= 1) mx = fmaxf(mx, __shfl_xor(mx, m, 16));
            float sum = 0.f;
#pragma unroll
            for (int ii = 0; ii < 16; ++ii) {
                float e = __expf(sc[ii] - mx);
                sh_alpha[r * 256 + l16 + 16 * ii] = e;
                sum += e;
            }
            for (int m = 8; m >= 1; m >>= 1) sum += __shfl_xor(sum, m, 16);
            float inv = 1.f / sum;
            __syncthreads();
            // ctx cols l16*8 .. +7 into inp cols 0..127
            float cacc[8];
#pragma unroll
            for (int q = 0; q < 8; ++q) cacc[q] = 0.f;
            for (int tq = 0; tq < 256; ++tq) {
                float al = sh_alpha[r * 256 + tq];
                const u16* vp = kr + (size_t)tq * 256 + 128 + l16 * 8;
                if (tq < 255) __builtin_prefetch(vp + 256, 0, 0);      // next V row
#pragma unroll
                for (int q = 0; q < 8; ++q) cacc[q] += al * bf2f(vp[q]);
            }
#pragma unroll
            for (int q = 0; q < 8; ++q) sh_inp[r * 160 + l16 * 8 + q] = f2bf(cacc[q] * inv);
        }
        __syncthreads();

        // 4. GRU cell GEMMs via WMMA
        v8f ai[3], ah[3];
#pragma unroll
        for (int g = 0; g < 3; ++g)
#pragma unroll
            for (int q = 0; q < 8; ++q) { ai[g][q] = 0.f; ah[g][q] = 0.f; }

#pragma unroll
        for (int kb = 0; kb < 5; ++kb) {   // K = 160 (ctx|y|pad)
            v16bf a = load_a_frag(sh_inp, 160, mt * 16, kb * 32, lane);
#pragma unroll
            for (int g = 0; g < 3; ++g) {
                v16bf bf = load_b_frag(cwih, 160, g * 128 + 16 * cg, kb * 32, lane);
                ai[g] = WMMA_BF16(a, bf, ai[g]);
            }
        }
#pragma unroll
        for (int kb = 0; kb < 4; ++kb) {   // K = 128
            v16bf a = load_a_frag(sh_hbf, 128, mt * 16, kb * 32, lane);
#pragma unroll
            for (int g = 0; g < 3; ++g) {
                v16bf bf = load_b_frag(cwhh, 128, g * 128 + 16 * cg, kb * 32, lane);
                ah[g] = WMMA_BF16(a, bf, ah[g]);
            }
        }
        __syncthreads();   // sh_hbf reads complete before overwrite

        // 5. gates
#pragma unroll
        for (int v = 0; v < 8; ++v) {
            int row = mt * 16 + v + 8 * hi;  // node index
            float hold = sh_hf[row * 128 + col];
            float r  = sigmoidf_(ai[0][v] + b_ir + ah[0][v] + b_hr);
            float zg = sigmoidf_(ai[1][v] + b_iz + ah[1][v] + b_hz);
            float nn2 = tanhf(ai[2][v] + b_in + r * (ah[2][v] + b_hn));
            float hnew = (1.f - zg) * nn2 + zg * hold;
            sh_hf[row * 128 + col] = hnew;
            sh_hbf[row * 128 + col] = f2bf(hnew);
        }
        __syncthreads();

        // 6. readout + output store
        if (tid < 512) {
            int i = tid >> 3, c = tid & 7;
            float acc = readb[c] + sh_yin[tid];
            for (int kk = 0; kk < 128; ++kk) acc += readW[c * 128 + kk] * sh_hf[i * 128 + kk];
            sh_ycur[tid] = acc;
            out[(((size_t)b * NN + i) * CC + c) * TOUT + t] = sh_xlast[tid] + acc;
        }
        __syncthreads();
    }
}

// ---------------------------------------------------------------------------
// Host launcher
// ---------------------------------------------------------------------------
extern "C" void kernel_launch(void* const* d_in, const int* in_sizes, int n_in,
                              void* d_out, int out_size, void* d_ws, size_t ws_size,
                              hipStream_t stream)
{
    (void)in_sizes; (void)n_in; (void)out_size; (void)ws_size;

    const float* X      = (const float*)d_in[0];
    const int*   phases = (const int*)  d_in[1];
    const float* gWih   = (const float*)d_in[2];
    const float* gWhh   = (const float*)d_in[3];
    const float* gbih   = (const float*)d_in[4];
    const float* gbhh   = (const float*)d_in[5];
    const float* ln_g   = (const float*)d_in[6];
    const float* ln_b   = (const float*)d_in[7];
    const float* tpw    = (const float*)d_in[8];
    const float* tpb    = (const float*)d_in[9];
    const float* S      = (const float*)d_in[10];
    const float* G      = (const float*)d_in[11];
    const float* Wself  = (const float*)d_in[12];
    const float* Wneigh = (const float*)d_in[13];
    const float* cWih   = (const float*)d_in[14];
    const float* cWhh   = (const float*)d_in[15];
    const float* cbih   = (const float*)d_in[16];
    const float* cbhh   = (const float*)d_in[17];
    const float* readW  = (const float*)d_in[18];
    const float* readb  = (const float*)d_in[19];
    const float* Wk     = (const float*)d_in[20];
    const float* Wv     = (const float*)d_in[21];
    const float* ltau   = (const float*)d_in[22];
    float* out = (float*)d_out;
    char* ws = (char*)d_ws;

    // workspace layout (bytes, 256-aligned)
    float* Wn    = (float*)(ws + 0);            //   73,728
    float* Afull = (float*)(ws + 73728);        //  524,288
    float* ut    = (float*)(ws + 598016);       //   16,384
    u16*   whh   = (u16*)(ws + 614400);         // 6,291,456
    u16*   wih   = (u16*)(ws + 6905856);        // 1,572,864
    u16*   cwih  = (u16*)(ws + 8478720);        //   122,880
    u16*   cwhh  = (u16*)(ws + 8601600);        //    98,304
    u16*   wkv   = (u16*)(ws + 8699904);        //    65,536
    u16*   wcat  = (u16*)(ws + 8765440);        //    81,920
    float* h0g   = (float*)(ws + 8847360);      // 1,048,576
    u16*   Hc    = (u16*)(ws + 9895936);        // 167,772,160
    u16*   HsHn  = (u16*)(ws + 177668096);      // 268,435,456 (reused as KV)
    u16*   Mm    = (u16*)(ws + 446103552);      // 134,217,728
    u16*   KV    = HsHn;                        // HsHn dead after mix_kernel

    prep_scalar_kernel<<<1, 256, 0, stream>>>(Wneigh, S, G, phases, tpw, tpb, Wn, Afull, ut);
    prep_weights_kernel<<<16080, 256, 0, stream>>>(gWih, gWhh, cWih, cWhh, Wk, Wv, Wself, Wn,
                                                   whh, wih, cwih, cwhh, wkv, wcat);
    gru_kernel<<<64, 256, 147968, stream>>>(X, gbih, gbhh, wih, whh, Hc);
    ln_kernel<<<65536, 256, 0, stream>>>(Hc, ln_g, ln_b, ut);
    // [Hs ; Hn] = Hc @ [W_self ; Wn]^T    (M = 524288, N = 256, K = 160)
    gemm_kernel<<<dim3(16384, 2), 256, 0, stream>>>(Hc, 160, wcat, 160, HsHn, 256, 160);
    mix_kernel<<<262144, 256, 0, stream>>>(HsHn, Afull, Mm, h0g);
    // [K ; V] = M @ [Wk ; Wv]^T           (M = 524288, N = 256, K = 128)
    gemm_kernel<<<dim3(16384, 2), 256, 0, stream>>>(Mm, 128, wkv, 128, KV, 256, 128);

    size_t shmem = (size_t)64 * 128 * 4 + 64 * 128 * 2 + 64 * 160 * 2 + 64 * 256 * 4 + 3 * 512 * 4;
    decoder_kernel<<<32, 1024, shmem, stream>>>(X, Afull, h0g, KV, cwih, cwhh,
                                                cbih, cbhh, readW, readb, ltau, out);
}
